// PMAT_42838003811021
// MI455X (gfx1250) — matrix operand order repaired and checked by reading, not compiled, not run on hardware
//
#include <hip/hip_runtime.h>
#include <hip/hip_bf16.h>
#include <stdint.h>

// ---------------------------------------------------------------------------
// Problem constants (from reference)
// ---------------------------------------------------------------------------
static constexpr int H_   = 512;
static constexpr int KC_  = 8192;
static constexpr int LQ_  = 4;
static constexpr int NM_  = 2;
static constexpr int SEQ_ = 64;
static constexpr int VD_  = 768;
static constexpr int TD_  = 768;
static constexpr int BN_  = 2048;

typedef __attribute__((ext_vector_type(16))) __bf16 bf16x16;
typedef __attribute__((ext_vector_type(8)))  __bf16 bf16x8;
typedef __attribute__((ext_vector_type(8)))  float  f32x8;
typedef __attribute__((ext_vector_type(4)))  float  f32x4;

// ---------------------------------------------------------------------------
// gfx1250 async global->LDS path (guarded; falls back to load + ds_store)
// Builtin signature (from compiler diagnostic / upstream def "vV4i*1V4i*3IiIi"):
//   void __builtin_amdgcn_global_load_async_to_lds_b128(
//        v4i addrspace(1)* src, v4i addrspace(3)* dst, imm int off, imm int cpol)
// ---------------------------------------------------------------------------
#if defined(__gfx1250__) && __has_builtin(__builtin_amdgcn_global_load_async_to_lds_b128)
#define ASYNC_LDS 1
#else
#define ASYNC_LDS 0
#endif

typedef __attribute__((__vector_size__(4 * sizeof(int)))) int v4i_;
#if ASYNC_LDS
typedef __attribute__((address_space(1))) v4i_* g_v4i_p;
typedef __attribute__((address_space(3))) v4i_* l_v4i_p;
#endif

__device__ __forceinline__ void g2l_async_b128(const __bf16* g, __bf16* l) {
#if ASYNC_LDS
  // generic LDS address truncates to the 32-bit LDS offset (aperture rules);
  // generic global address equals its AS1 value.
  __builtin_amdgcn_global_load_async_to_lds_b128(
      (g_v4i_p)(uintptr_t)g,
      (l_v4i_p)(unsigned)(uintptr_t)l,
      0, 0);
#else
  (void)g; (void)l;
#endif
}

__device__ __forceinline__ void wait_async0() {
#if ASYNC_LDS
#if __has_builtin(__builtin_amdgcn_s_wait_asynccnt)
  __builtin_amdgcn_s_wait_asynccnt(0);
#else
  asm volatile("s_wait_asynccnt 0x0" ::: "memory");
#endif
#endif
}

// ---------------------------------------------------------------------------
// WMMA GEMM:  C[M x N] = A1[M x K1](f32) * W1[N x K1](bf16)^T
//                       (+ A2[M x K2](f32) * W2[N x K2](bf16)^T)  + bias[N]
// Tile: 128(M) x 128(N), K-chunk 32, 256 threads = 8 waves.
// Wave tile 32(M) x 64(N): 2 A frags reused over 4 B frags -> 8 WMMA/chunk.
// ---------------------------------------------------------------------------
#define TM 128
#define TN 128
#define TK 32
#define LDT 48  // padded LDS row stride (keeps 16B alignment, de-phases banks)

__global__ __launch_bounds__(256) void k_gemm(
    const float* __restrict__ A1, long lda1, const __bf16* __restrict__ W1, int K1,
    const float* __restrict__ A2, long lda2, const __bf16* __restrict__ W2, int K2,
    const float* __restrict__ bias,
    float* __restrict__ C, long ldc, int M, int N)
{
  __shared__ __attribute__((aligned(16))) __bf16 As[TM][LDT];
  __shared__ __attribute__((aligned(16))) __bf16 Bs[TN][LDT];

  const int tid  = threadIdx.x;
  const int lane = tid & 31;
  const int wv   = tid >> 5;
  const int wm   = (wv & 3) * 32;  // wave tile row inside block tile
  const int wn   = (wv >> 2) * 64; // wave tile col inside block tile
  const int lr   = lane & 15;
  const int kh   = lane >> 4;
  const int bm   = blockIdx.y * TM;
  const int bn   = blockIdx.x * TN;

  f32x8 acc[2][4];
#pragma unroll
  for (int i = 0; i < 2; ++i)
#pragma unroll
    for (int j = 0; j < 4; ++j)
#pragma unroll
      for (int e = 0; e < 8; ++e) acc[i][j][e] = 0.0f;

  const int c1 = K1 >> 5;
  const int cT = c1 + (K2 >> 5);

  for (int c = 0; c < cT; ++c) {
    const float* Ap; long lda; const __bf16* Wp; int Kd; int k0;
    if (c < c1) { Ap = A1; lda = lda1; Wp = W1; Kd = K1; k0 = c * TK; }
    else        { Ap = A2; lda = lda2; Wp = W2; Kd = K2; k0 = (c - c1) * TK; }

    // ---- stage A: 128x32 f32 -> bf16 in LDS (16 elems / thread) ----
    {
      const int r  = tid >> 1;
      const int hb = (tid & 1) * 16;
      const int gm = bm + r;
      __bf16 tmp[16];
      if (gm < M) {
        const float* src = Ap + (size_t)gm * lda + k0 + hb;
        __builtin_prefetch(src + TK, 0, 1);  // global_prefetch_b8 toward next chunk
#pragma unroll
        for (int q = 0; q < 4; ++q) {
          f32x4 v = *(const f32x4*)(src + q * 4);
          tmp[q * 4 + 0] = (__bf16)v[0];
          tmp[q * 4 + 1] = (__bf16)v[1];
          tmp[q * 4 + 2] = (__bf16)v[2];
          tmp[q * 4 + 3] = (__bf16)v[3];
        }
      } else {
#pragma unroll
        for (int q = 0; q < 16; ++q) tmp[q] = (__bf16)0.0f;
      }
      bf16x8 lo, hi;
#pragma unroll
      for (int e = 0; e < 8; ++e) { lo[e] = tmp[e]; hi[e] = tmp[e + 8]; }
      *(bf16x8*)&As[r][hb]     = lo;
      *(bf16x8*)&As[r][hb + 8] = hi;
    }
    // ---- stage B: 128x32 bf16 weights (async LDS DMA when available) ----
    {
      const int r  = tid >> 1;
      const int qb = (tid & 1) * 16;
      const int gn = bn + r;
      if (gn < N) {
        const __bf16* src = Wp + (size_t)gn * Kd + k0 + qb;
#if ASYNC_LDS
        g2l_async_b128(src,     &Bs[r][qb]);
        g2l_async_b128(src + 8, &Bs[r][qb + 8]);
#else
        bf16x8 v0 = *(const bf16x8*)src;
        bf16x8 v1 = *(const bf16x8*)(src + 8);
        *(bf16x8*)&Bs[r][qb]     = v0;
        *(bf16x8*)&Bs[r][qb + 8] = v1;
#endif
      } else {
        bf16x8 z;
#pragma unroll
        for (int e = 0; e < 8; ++e) z[e] = (__bf16)0.0f;
        *(bf16x8*)&Bs[r][qb]     = z;
        *(bf16x8*)&Bs[r][qb + 8] = z;
      }
    }
    wait_async0();
    __syncthreads();

    // ---- fragments: contiguous k order k = 16*kh + s for both A and B ----
    bf16x16 fa[2], fb[4];
#pragma unroll
    for (int i = 0; i < 2; ++i) {
      // A 16x32: lane group kh holds K = kh*16..+15 contiguous
      const __bf16* pr = &As[wm + i * 16 + lr][0];
      bf16x8 lo = *(const bf16x8*)(pr + kh * 16);
      bf16x8 hi = *(const bf16x8*)(pr + kh * 16 + 8);
#pragma unroll
      for (int e = 0; e < 8; ++e) { fa[i][e] = lo[e]; fa[i][e + 8] = hi[e]; }
    }
#pragma unroll
    for (int j = 0; j < 4; ++j) {
      // B 32x16: lane group kh holds K = kh*16..+15 contiguous, N = lane&15
      const __bf16* pr = &Bs[wn + j * 16 + lr][kh * 16];
      bf16x8 lo = *(const bf16x8*)(pr);
      bf16x8 hi = *(const bf16x8*)(pr + 8);
#pragma unroll
      for (int e = 0; e < 8; ++e) { fb[j][e] = lo[e]; fb[j][e + 8] = hi[e]; }
    }

#pragma unroll
    for (int i = 0; i < 2; ++i)
#pragma unroll
      for (int j = 0; j < 4; ++j)
        acc[i][j] = __builtin_amdgcn_wmma_f32_16x16x32_bf16(
            false, fa[i], false, fb[j], (short)0, acc[i][j], false, false);

    __syncthreads();
  }

  // ---- epilogue: C/D layout VGPR r -> M = r + 8*kh, N = lane&15 ----
#pragma unroll
  for (int j = 0; j < 4; ++j) {
    const int gn = bn + wn + j * 16 + lr;
    if (gn >= N) continue;
    const float bv = bias ? bias[gn] : 0.0f;
#pragma unroll
    for (int i = 0; i < 2; ++i) {
#pragma unroll
      for (int r = 0; r < 8; ++r) {
        const int gm = bm + wm + i * 16 + kh * 8 + r;
        if (gm < M) C[(size_t)gm * ldc + gn] = acc[i][j][r] + bv;
      }
    }
  }
}

// ---------------------------------------------------------------------------
// Elementwise / reduction kernels
// ---------------------------------------------------------------------------
__global__ void k_cvt_bf16(const float* __restrict__ x, __bf16* __restrict__ y, long n) {
  for (long i = (long)blockIdx.x * blockDim.x + threadIdx.x; i < n;
       i += (long)gridDim.x * blockDim.x)
    y[i] = (__bf16)x[i];
}

__global__ void k_zero(float* __restrict__ p, long n) {
  for (long i = (long)blockIdx.x * blockDim.x + threadIdx.x; i < n;
       i += (long)gridDim.x * blockDim.x)
    p[i] = 0.0f;
}

__global__ void k_add2(const float* __restrict__ a, const float* __restrict__ b,
                       float* __restrict__ y, int n) {
  int i = blockIdx.x * blockDim.x + threadIdx.x;
  if (i < n) y[i] = a[i] + b[i];
}

__device__ __forceinline__ float sigm(float x) { return 1.0f / (1.0f + expf(-x)); }

__global__ void k_lstm_cell(const float* __restrict__ gates, float* __restrict__ h,
                            float* __restrict__ c) {
  long i = (long)blockIdx.x * blockDim.x + threadIdx.x;
  if (i >= (long)BN_ * H_) return;
  long b = i / H_;
  int  j = (int)(i % H_);
  const float* g = gates + (size_t)b * 4 * H_;
  float gi = g[j], gf = g[H_ + j], gg = g[2 * H_ + j], go = g[3 * H_ + j];
  float cn = sigm(gf) * c[i] + sigm(gi) * tanhf(gg);
  c[i] = cn;
  h[i] = sigm(go) * tanhf(cn);
}

__device__ __forceinline__ float gelu_exact(float x) {
  return 0.5f * x * (1.0f + erff(x * 0.70710678118654752f));
}

// LayerNorm (+ optional relu=1 / gelu=2) on rows of width H_=512; in-place OK.
__global__ __launch_bounds__(256) void k_ln_act(const float* __restrict__ x,
                                                const float* __restrict__ g,
                                                const float* __restrict__ bta,
                                                float* __restrict__ y, int act) {
  __shared__ float red[256];
  const int b = blockIdx.x, t = threadIdx.x;
  const float* xr = x + (size_t)b * H_;
  float v0 = xr[t], v1 = xr[t + 256];

  red[t] = v0 + v1; __syncthreads();
  for (int o = 128; o > 0; o >>= 1) { if (t < o) red[t] += red[t + o]; __syncthreads(); }
  float mean = red[0] * (1.0f / H_); __syncthreads();

  float d0 = v0 - mean, d1 = v1 - mean;
  red[t] = d0 * d0 + d1 * d1; __syncthreads();
  for (int o = 128; o > 0; o >>= 1) { if (t < o) red[t] += red[t + o]; __syncthreads(); }
  float inv = rsqrtf(red[0] * (1.0f / H_) + 1e-5f);

  float y0 = d0 * inv * g[t]       + bta[t];
  float y1 = d1 * inv * g[t + 256] + bta[t + 256];
  if (act == 1) { y0 = fmaxf(y0, 0.0f); y1 = fmaxf(y1, 0.0f); }
  else if (act == 2) { y0 = gelu_exact(y0); y1 = gelu_exact(y1); }
  y[(size_t)b * H_ + t] = y0;
  y[(size_t)b * H_ + t + 256] = y1;
}

// logits_pref = p @ W2^T + b2 ; w = softmax(logits / T) (NM_=2, renorm is no-op)
__global__ __launch_bounds__(256) void k_pref_head(const float* __restrict__ p,
                                                   const float* __restrict__ W2,
                                                   const float* __restrict__ b2,
                                                   const float* __restrict__ temp,
                                                   float* __restrict__ wmix) {
  __shared__ float r0[256], r1[256];
  const int b = blockIdx.x, t = threadIdx.x;
  const float* pr = p + (size_t)b * H_;
  r0[t] = pr[t] * W2[t] + pr[t + 256] * W2[t + 256];
  r1[t] = pr[t] * W2[H_ + t] + pr[t + 256] * W2[H_ + t + 256];
  __syncthreads();
  for (int o = 128; o > 0; o >>= 1) {
    if (t < o) { r0[t] += r0[t + o]; r1[t] += r1[t + o]; }
    __syncthreads();
  }
  if (t == 0) {
    float T  = temp[0];
    float l0 = (r0[0] + b2[0]) / T;
    float l1 = (r1[0] + b2[1]) / T;
    float m  = fmaxf(l0, l1);
    float e0 = expf(l0 - m), e1 = expf(l1 - m);
    float s  = e0 + e1;
    wmix[2 * b]     = e0 / s;
    wmix[2 * b + 1] = e1 / s;
  }
}

__global__ void k_fuse(const float* __restrict__ tx, const float* __restrict__ vs,
                       const float* __restrict__ w, float* __restrict__ o) {
  long i = (long)blockIdx.x * blockDim.x + threadIdx.x;
  if (i >= (long)BN_ * H_) return;
  long b = i / H_;
  o[i] = w[2 * b] * tx[i] + w[2 * b + 1] * vs[i];
}

// Per-codeword: convert row to bf16 and compute ||c||^2
__global__ __launch_bounds__(256) void k_cb_prep(const float* __restrict__ cb,
                                                 __bf16* __restrict__ cbb,
                                                 float* __restrict__ cbn) {
  __shared__ float red[256];
  const int j = blockIdx.x, t = threadIdx.x;
  const float* row = cb + (size_t)j * H_;
  float v0 = row[t], v1 = row[t + 256];
  cbb[(size_t)j * H_ + t]       = (__bf16)v0;
  cbb[(size_t)j * H_ + t + 256] = (__bf16)v1;
  red[t] = v0 * v0 + v1 * v1; __syncthreads();
  for (int o = 128; o > 0; o >>= 1) { if (t < o) red[t] += red[t + o]; __syncthreads(); }
  if (t == 0) cbn[j] = red[0];
}

// In-place: dots -> -sqrt(max(||r||^2 - 2*dot + ||c||^2, 0)), plus argmin row id
__global__ __launch_bounds__(256) void k_rvq_finalize(const float* __restrict__ resid,
                                                      const float* __restrict__ cbn,
                                                      float* __restrict__ dots, long ldc,
                                                      int* __restrict__ ids) {
  __shared__ float red[256];
  __shared__ float bd[256];
  __shared__ int   bi[256];
  const int b = blockIdx.x, t = threadIdx.x;
  const float* rr = resid + (size_t)b * H_;
  float v0 = rr[t], v1 = rr[t + 256];
  red[t] = v0 * v0 + v1 * v1; __syncthreads();
  for (int o = 128; o > 0; o >>= 1) { if (t < o) red[t] += red[t + o]; __syncthreads(); }
  const float rn = red[0];
  __syncthreads();

  float best = 3.4e38f;
  int   bidx = 0;
  float* drow = dots + (size_t)b * ldc;
  for (int j = t; j < KC_; j += 256) {
    float d2   = rn - 2.0f * drow[j] + cbn[j];
    float dist = sqrtf(fmaxf(d2, 0.0f));
    drow[j] = -dist;
    if (dist < best) { best = dist; bidx = j; }
  }
  bd[t] = best; bi[t] = bidx; __syncthreads();
  for (int o = 128; o > 0; o >>= 1) {
    if (t < o) {
      if (bd[t + o] < bd[t] || (bd[t + o] == bd[t] && bi[t + o] < bi[t])) {
        bd[t] = bd[t + o]; bi[t] = bi[t + o];
      }
    }
    __syncthreads();
  }
  if (t == 0) ids[b] = bi[0];
}

__global__ void k_rvq_update(const float* __restrict__ cbl, const int* __restrict__ ids,
                             float* __restrict__ qsum, float* __restrict__ resid) {
  long i = (long)blockIdx.x * blockDim.x + threadIdx.x;
  if (i >= (long)BN_ * H_) return;
  long b = i / H_;
  int  j = (int)(i % H_);
  float q = cbl[(size_t)ids[b] * H_ + j];
  qsum[i]  += q;
  resid[i] -= q;
}

// ---------------------------------------------------------------------------
// Orchestration
// ---------------------------------------------------------------------------
extern "C" void kernel_launch(void* const* d_in, const int* in_sizes, int n_in,
                              void* d_out, int out_size, void* d_ws, size_t ws_size,
                              hipStream_t stream) {
  (void)in_sizes; (void)n_in; (void)out_size; (void)ws_size;

  const float* text_feat    = (const float*)d_in[0];
  const float* vision_feat  = (const float*)d_in[1];
  const float* user_history = (const float*)d_in[2];
  const float* l0_Wih = (const float*)d_in[3];
  const float* l0_Whh = (const float*)d_in[4];
  const float* l0_bih = (const float*)d_in[5];
  const float* l0_bhh = (const float*)d_in[6];
  const float* l1_Wih = (const float*)d_in[7];
  const float* l1_Whh = (const float*)d_in[8];
  const float* l1_bih = (const float*)d_in[9];
  const float* l1_bhh = (const float*)d_in[10];
  const float* pref_W1   = (const float*)d_in[11];
  const float* pref_b1   = (const float*)d_in[12];
  const float* pref_ln_g = (const float*)d_in[13];
  const float* pref_ln_b = (const float*)d_in[14];
  const float* pref_W2   = (const float*)d_in[15];
  const float* pref_b2   = (const float*)d_in[16];
  const float* temperature = (const float*)d_in[17];
  const float* txt_W   = (const float*)d_in[18];
  const float* txt_b   = (const float*)d_in[19];
  const float* txt_ln_g = (const float*)d_in[20];
  const float* txt_ln_b = (const float*)d_in[21];
  const float* vis_W   = (const float*)d_in[22];
  const float* vis_b   = (const float*)d_in[23];
  const float* vis_ln_g = (const float*)d_in[24];
  const float* vis_ln_b = (const float*)d_in[25];
  const float* fuse_W   = (const float*)d_in[26];
  const float* fuse_b   = (const float*)d_in[27];
  const float* fuse_ln_g = (const float*)d_in[28];
  const float* fuse_ln_b = (const float*)d_in[29];
  const float* proj_W   = (const float*)d_in[30];
  const float* proj_b   = (const float*)d_in[31];
  const float* codebooks = (const float*)d_in[32];

  float* out = (float*)d_out;

  // -------- workspace carve --------
  char* wp = (char*)d_ws;
  auto carve = [&](size_t bytes) -> void* {
    void* p = wp; wp += (bytes + 255) & ~(size_t)255; return p;
  };
  __bf16* w0ih_bf = (__bf16*)carve((size_t)4 * H_ * H_ * sizeof(__bf16));
  __bf16* w0hh_bf = (__bf16*)carve((size_t)4 * H_ * H_ * sizeof(__bf16));
  __bf16* w1ih_bf = (__bf16*)carve((size_t)4 * H_ * H_ * sizeof(__bf16));
  __bf16* w1hh_bf = (__bf16*)carve((size_t)4 * H_ * H_ * sizeof(__bf16));
  __bf16* wpref_bf = (__bf16*)carve((size_t)H_ * H_ * sizeof(__bf16));
  __bf16* wtxt_bf  = (__bf16*)carve((size_t)H_ * TD_ * sizeof(__bf16));
  __bf16* wvis_bf  = (__bf16*)carve((size_t)H_ * VD_ * sizeof(__bf16));
  __bf16* wfuse_bf = (__bf16*)carve((size_t)H_ * H_ * sizeof(__bf16));
  __bf16* wproj_bf = (__bf16*)carve((size_t)H_ * H_ * sizeof(__bf16));
  __bf16* cb_bf    = (__bf16*)carve((size_t)KC_ * H_ * sizeof(__bf16));
  float* bias0 = (float*)carve((size_t)4 * H_ * sizeof(float));
  float* bias1 = (float*)carve((size_t)4 * H_ * sizeof(float));
  float* h0 = (float*)carve((size_t)BN_ * H_ * sizeof(float));
  float* c0 = (float*)carve((size_t)BN_ * H_ * sizeof(float));
  float* h1 = (float*)carve((size_t)BN_ * H_ * sizeof(float));
  float* c1 = (float*)carve((size_t)BN_ * H_ * sizeof(float));
  float* gates = (float*)carve((size_t)BN_ * 4 * H_ * sizeof(float));
  float* pbuf  = (float*)carve((size_t)BN_ * H_ * sizeof(float));
  float* txtbf = (float*)carve((size_t)BN_ * H_ * sizeof(float));
  float* visbf = (float*)carve((size_t)BN_ * H_ * sizeof(float));
  float* fubuf = (float*)carve((size_t)BN_ * H_ * sizeof(float));
  float* feat  = (float*)carve((size_t)BN_ * H_ * sizeof(float));  // = residual
  float* wmix  = (float*)carve((size_t)BN_ * NM_ * sizeof(float));
  float* cbn   = (float*)carve((size_t)KC_ * sizeof(float));
  int*   ids   = (int*)carve((size_t)BN_ * sizeof(int));

  auto gs = [](long n) { return (unsigned)((n + 255) / 256); };

  // -------- weight prep --------
  k_cvt_bf16<<<gs(4L * H_ * H_), 256, 0, stream>>>(l0_Wih, w0ih_bf, 4L * H_ * H_);
  k_cvt_bf16<<<gs(4L * H_ * H_), 256, 0, stream>>>(l0_Whh, w0hh_bf, 4L * H_ * H_);
  k_cvt_bf16<<<gs(4L * H_ * H_), 256, 0, stream>>>(l1_Wih, w1ih_bf, 4L * H_ * H_);
  k_cvt_bf16<<<gs(4L * H_ * H_), 256, 0, stream>>>(l1_Whh, w1hh_bf, 4L * H_ * H_);
  k_cvt_bf16<<<gs((long)H_ * H_), 256, 0, stream>>>(pref_W1, wpref_bf, (long)H_ * H_);
  k_cvt_bf16<<<gs((long)H_ * TD_), 256, 0, stream>>>(txt_W, wtxt_bf, (long)H_ * TD_);
  k_cvt_bf16<<<gs((long)H_ * VD_), 256, 0, stream>>>(vis_W, wvis_bf, (long)H_ * VD_);
  k_cvt_bf16<<<gs((long)H_ * H_), 256, 0, stream>>>(fuse_W, wfuse_bf, (long)H_ * H_);
  k_cvt_bf16<<<gs((long)H_ * H_), 256, 0, stream>>>(proj_W, wproj_bf, (long)H_ * H_);
  k_add2<<<gs(4 * H_), 256, 0, stream>>>(l0_bih, l0_bhh, bias0, 4 * H_);
  k_add2<<<gs(4 * H_), 256, 0, stream>>>(l1_bih, l1_bhh, bias1, 4 * H_);

  // -------- LSTM state init --------
  k_zero<<<gs((long)BN_ * H_), 256, 0, stream>>>(h0, (long)BN_ * H_);
  k_zero<<<gs((long)BN_ * H_), 256, 0, stream>>>(c0, (long)BN_ * H_);
  k_zero<<<gs((long)BN_ * H_), 256, 0, stream>>>(h1, (long)BN_ * H_);
  k_zero<<<gs((long)BN_ * H_), 256, 0, stream>>>(c1, (long)BN_ * H_);

  // -------- 2-layer LSTM, layers interleaved per timestep --------
  const dim3 ggate((4 * H_) / TN, BN_ / TM);
  for (int t = 0; t < SEQ_; ++t) {
    // layer 0: gates = x_t @ Wih^T + h0 @ Whh^T + (bih+bhh)
    k_gemm<<<ggate, 256, 0, stream>>>(
        user_history + (size_t)t * H_, (long)SEQ_ * H_, w0ih_bf, H_,
        h0, (long)H_, w0hh_bf, H_,
        bias0, gates, (long)4 * H_, BN_, 4 * H_);
    k_lstm_cell<<<gs((long)BN_ * H_), 256, 0, stream>>>(gates, h0, c0);
    // layer 1: gates = h0 @ Wih^T + h1 @ Whh^T + (bih+bhh)
    k_gemm<<<ggate, 256, 0, stream>>>(
        h0, (long)H_, w1ih_bf, H_,
        h1, (long)H_, w1hh_bf, H_,
        bias1, gates, (long)4 * H_, BN_, 4 * H_);
    k_lstm_cell<<<gs((long)BN_ * H_), 256, 0, stream>>>(gates, h1, c1);
  }

  // -------- preference head --------
  const dim3 gHH(H_ / TN, BN_ / TM);
  k_gemm<<<gHH, 256, 0, stream>>>(h1, (long)H_, wpref_bf, H_,
                                  nullptr, 0, nullptr, 0,
                                  pref_b1, pbuf, (long)H_, BN_, H_);
  k_ln_act<<<BN_, 256, 0, stream>>>(pbuf, pref_ln_g, pref_ln_b, pbuf, 1);
  k_pref_head<<<BN_, 256, 0, stream>>>(pbuf, pref_W2, pref_b2, temperature, wmix);

  // -------- modality encoders --------
  k_gemm<<<gHH, 256, 0, stream>>>(text_feat, (long)TD_, wtxt_bf, TD_,
                                  nullptr, 0, nullptr, 0,
                                  txt_b, txtbf, (long)H_, BN_, H_);
  k_ln_act<<<BN_, 256, 0, stream>>>(txtbf, txt_ln_g, txt_ln_b, txtbf, 1);
  k_gemm<<<gHH, 256, 0, stream>>>(vision_feat, (long)VD_, wvis_bf, VD_,
                                  nullptr, 0, nullptr, 0,
                                  vis_b, visbf, (long)H_, BN_, H_);
  k_ln_act<<<BN_, 256, 0, stream>>>(visbf, vis_ln_g, vis_ln_b, visbf, 1);

  // -------- fusion + projection --------
  k_fuse<<<gs((long)BN_ * H_), 256, 0, stream>>>(txtbf, visbf, wmix, fubuf);
  k_gemm<<<gHH, 256, 0, stream>>>(fubuf, (long)H_, wfuse_bf, H_,
                                  nullptr, 0, nullptr, 0,
                                  fuse_b, pbuf, (long)H_, BN_, H_);
  k_ln_act<<<BN_, 256, 0, stream>>>(pbuf, fuse_ln_g, fuse_ln_b, pbuf, 2);  // gelu
  k_gemm<<<gHH, 256, 0, stream>>>(pbuf, (long)H_, wproj_bf, H_,
                                  nullptr, 0, nullptr, 0,
                                  proj_b, feat, (long)H_, BN_, H_);

  // -------- residual VQ (4 levels) --------
  float* qsum = out + (size_t)BN_ * LQ_ * KC_;
  k_zero<<<gs((long)BN_ * H_), 256, 0, stream>>>(qsum, (long)BN_ * H_);

  const dim3 grvq(KC_ / TN, BN_ / TM);
  for (int lvl = 0; lvl < LQ_; ++lvl) {
    const float* cbl = codebooks + (size_t)lvl * KC_ * H_;
    float* logits = out + (size_t)lvl * KC_;  // row stride LQ_*KC_
    k_cb_prep<<<KC_, 256, 0, stream>>>(cbl, cb_bf, cbn);
    k_gemm<<<grvq, 256, 0, stream>>>(feat, (long)H_, cb_bf, H_,
                                     nullptr, 0, nullptr, 0,
                                     nullptr, logits, (long)LQ_ * KC_, BN_, KC_);
    k_rvq_finalize<<<BN_, 256, 0, stream>>>(feat, cbn, logits, (long)LQ_ * KC_, ids);
    k_rvq_update<<<gs((long)BN_ * H_), 256, 0, stream>>>(cbl, ids, qsum, feat);
  }
}
